// BitnetFeedForward_14611478741500
// MI455X (gfx1250) — 8-bit matrix operand order repaired and checked by reading, compiled, not run on hardware
//
#include <hip/hip_runtime.h>
#include <hip/hip_bf16.h>


typedef __attribute__((ext_vector_type(8))) int v8i;
typedef __attribute__((ext_vector_type(4))) int v4i_t;

#define DIM    2048
#define INNER  8192
#define NTOK   8192
#define EPSF   1e-5f

#if defined(__HIP_DEVICE_COMPILE__) && __has_builtin(__builtin_amdgcn_global_load_async_to_lds_b128)
#define HAS_ASYNC_LDS 1
#else
#define HAS_ASYNC_LDS 0
#endif

#if HAS_ASYNC_LDS
typedef __attribute__((address_space(1))) v4i_t* gptr_b128;
typedef __attribute__((address_space(3))) v4i_t* lptr_b128;
#endif

__device__ __forceinline__ void wait_async_lds() {
#if HAS_ASYNC_LDS
#if __has_builtin(__builtin_amdgcn_s_wait_asynccnt)
    __builtin_amdgcn_s_wait_asynccnt(0);
#else
    asm volatile("s_wait_asynccnt 0" ::: "memory");
#endif
#endif
}

__device__ __forceinline__ float gelu_exact(float x) {
    return 0.5f * x * (1.0f + erff(x * 0.70710678118654752440f));
}

__device__ __forceinline__ signed char quant_ternary(float v) {
    int q = (int)rintf(v);
    q = q > 1 ? 1 : (q < -1 ? -1 : q);
    return (signed char)q;
}

__device__ __forceinline__ signed char quant_i8(float v) {
    int q = (int)rintf(v);
    q = q > 127 ? 127 : (q < -128 ? -128 : q);
    return (signed char)q;
}


__global__ void zero_f32(float* p, int n) {
    int i = blockIdx.x * blockDim.x + threadIdx.x;
    if (i < n) p[i] = 0.0f;
}

__global__ __launch_bounds__(256) void abs_sum_kernel(const float* __restrict__ W,
                                                      long n, float* __restrict__ acc) {
    __shared__ float red[256];
    float s = 0.0f;
    long stride = (long)gridDim.x * blockDim.x;
    for (long i = blockIdx.x * (long)blockDim.x + threadIdx.x; i < n; i += stride)
        s += fabsf(W[i]);
    red[threadIdx.x] = s;
    __syncthreads();
    for (int k = 128; k > 0; k >>= 1) {
        if ((int)threadIdx.x < k) red[threadIdx.x] += red[threadIdx.x + k];
        __syncthreads();
    }
    if (threadIdx.x == 0) atomicAdd(acc, red[0]);
}

__global__ void weight_scale_kernel(const float* __restrict__ wsum,
                                    float* __restrict__ scales,
                                    float n1, float n2) {
    if (threadIdx.x == 0) {
        scales[0] = 1.0f / fmaxf(wsum[0] / n1, EPSF);
        scales[1] = 1.0f / fmaxf(wsum[1] / n2, EPSF);
    }
}

__global__ __launch_bounds__(256) void weight_quant_kernel(const float* __restrict__ W,
                                                           signed char* __restrict__ Q,
                                                           const float* __restrict__ scalePtr,
                                                           long n4) {
    const float s = *scalePtr;
    long stride = (long)gridDim.x * blockDim.x;
    for (long i = blockIdx.x * (long)blockDim.x + threadIdx.x; i < n4; i += stride) {
        float4 v = ((const float4*)W)[i];
        char4 o;
        o.x = quant_ternary(v.x * s);
        o.y = quant_ternary(v.y * s);
        o.z = quant_ternary(v.z * s);
        o.w = quant_ternary(v.w * s);
        ((char4*)Q)[i] = o;
    }
}

__global__ __launch_bounds__(256) void rms_act_quant_kernel(const float* __restrict__ X,
                                                            signed char* __restrict__ Q,
                                                            float* __restrict__ aScale,
                                                            int rowLen) {
    __shared__ float sh[INNER];
    __shared__ float redS[256];
    __shared__ float redM[256];

    const long t = blockIdx.x;
    const float* row = X + t * (long)rowLen;
    const int n4 = rowLen >> 2;

    float ss = 0.0f, amax = 0.0f;
    for (int i = threadIdx.x; i < n4; i += 256) {
        float4 v = ((const float4*)row)[i];
        ((float4*)sh)[i] = v;
        ss += v.x * v.x + v.y * v.y + v.z * v.z + v.w * v.w;
        amax = fmaxf(amax, fmaxf(fmaxf(fabsf(v.x), fabsf(v.y)),
                                 fmaxf(fabsf(v.z), fabsf(v.w))));
    }
    redS[threadIdx.x] = ss;
    redM[threadIdx.x] = amax;
    __syncthreads();
    for (int k = 128; k > 0; k >>= 1) {
        if ((int)threadIdx.x < k) {
            redS[threadIdx.x] += redS[threadIdx.x + k];
            redM[threadIdx.x] = fmaxf(redM[threadIdx.x], redM[threadIdx.x + k]);
        }
        __syncthreads();
    }
    const float norm = fmaxf(sqrtf(redS[0]), 1e-12f);
    const float mul  = sqrtf((float)rowLen) / norm;
    const float scale = 127.0f / fmaxf(redM[0] * mul, EPSF);
    if (threadIdx.x == 0) aScale[t] = scale;
    const float qs = mul * scale;

    signed char* qrow = Q + t * (long)rowLen;
    for (int i = threadIdx.x; i < n4; i += 256) {
        float4 v = ((float4*)sh)[i];
        char4 o;
        o.x = quant_i8(v.x * qs);
        o.y = quant_i8(v.y * qs);
        o.z = quant_i8(v.z * qs);
        o.w = quant_i8(v.w * qs);
        ((char4*)qrow)[i] = o;
    }
}

template <bool GELU_EPILOGUE>
__global__ __launch_bounds__(256) void gemm_i8_wmma_kernel(
        const signed char* __restrict__ Aq,
        const signed char* __restrict__ Bq,
        const float* __restrict__ aScale,
        const float* __restrict__ wScalePtr,
        float* __restrict__ Cout,
        int M, int N, int K) {
    __shared__ alignas(16) signed char As[2][128 * 64];
    __shared__ alignas(16) signed char Bs[2][128 * 64];

    const int tid   = threadIdx.x;
    const int lane  = tid & 31;
    const int wid   = tid >> 5;
    const int waveM = wid >> 1;
    const int waveN = wid & 1;
    const int lrow  = lane & 15;
    const int kHi   = lane >> 4;

    const long mBase = (long)blockIdx.y * 128;
    const long nBase = (long)blockIdx.x * 128;

    auto stage = [&](int buf, int k0) {
#pragma unroll
        for (int i = 0; i < 2; ++i) {
            const int idx = tid + i * 256;
            const int row = idx >> 2;
            const int col = (idx & 3) * 16;
            const long aOff = (mBase + row) * (long)K + k0 + col;
            const long bOff = (nBase + row) * (long)K + k0 + col;
#if HAS_ASYNC_LDS
            __builtin_amdgcn_global_load_async_to_lds_b128(
                (gptr_b128)(Aq + aOff), (lptr_b128)&As[buf][row * 64 + col],
                 0,  0);
            __builtin_amdgcn_global_load_async_to_lds_b128(
                (gptr_b128)(Bq + bOff), (lptr_b128)&Bs[buf][row * 64 + col],
                 0,  0);
#else
            *((int4*)&As[buf][row * 64 + col]) = *((const int4*)(Aq + aOff));
            *((int4*)&Bs[buf][row * 64 + col]) = *((const int4*)(Bq + bOff));
#endif
        }
    };

    v8i acc[2][4] = {};

    stage(0, 0);
    wait_async_lds();
    __syncthreads();

    int buf = 0;
    for (int k0 = 0; k0 < K; k0 += 64) {
        if (k0 + 64 < K) stage(buf ^ 1, k0 + 64);

        v8i afr[2], bfr[4];
#pragma unroll
        for (int tm = 0; tm < 2; ++tm) {
            const signed char* ap = &As[buf][(waveM * 32 + tm * 16 + lrow) * 64];
#pragma unroll
            for (int v = 0; v < 8; ++v) {
                const int k = ((v >> 2) << 5) + (kHi << 4) + ((v & 3) << 2);
                afr[tm][v] = *(const int*)(ap + k);
            }
        }
#pragma unroll
        for (int tn = 0; tn < 4; ++tn) {
            const signed char* bp = &Bs[buf][(waveN * 64 + tn * 16 + lrow) * 64];
#pragma unroll
            for (int v = 0; v < 8; ++v) {
                const int k = ((v >> 2) << 5) + (kHi << 4) + ((v & 3) << 2);
                bfr[tn][v] = *(const int*)(bp + k);
            }
        }

#pragma unroll
        for (int tm = 0; tm < 2; ++tm)
#pragma unroll
            for (int tn = 0; tn < 4; ++tn)
                acc[tm][tn] = __builtin_amdgcn_wmma_i32_16x16x64_iu8(
                     true, afr[tm],  true, bfr[tn],
                    acc[tm][tn],  false,  false);

        wait_async_lds();
        __syncthreads();
        buf ^= 1;
    }

    const float sw = *wScalePtr;
#pragma unroll
    for (int tm = 0; tm < 2; ++tm) {
        const long mTile = mBase + waveM * 32 + tm * 16;
#pragma unroll
        for (int tn = 0; tn < 4; ++tn) {
            const long n = nBase + waveN * 64 + tn * 16 + lrow;
#pragma unroll
            for (int r = 0; r < 8; ++r) {
                const long m = mTile + r + kHi * 8;
                float v = (float)acc[tm][tn][r] / (aScale[m] * sw);
                if (GELU_EPILOGUE) v = gelu_exact(gelu_exact(v));
                Cout[m * (long)N + n] = v;
            }
        }
    }
}


extern "C" void kernel_launch(void* const* d_in, const int* in_sizes, int n_in,
                              void* d_out, int out_size, void* d_ws, size_t ws_size,
                              hipStream_t stream) {
    const float* x  = (const float*)d_in[0];
    const float* w1 = (const float*)d_in[1];
    const float* w2 = (const float*)d_in[2];
    float* out = (float*)d_out;

    const long nW1 = (long)INNER * DIM;
    const long nW2 = (long)DIM * INNER;
    const long nX  = (long)NTOK * DIM;
    const long nH  = (long)NTOK * INNER;

    char* ws = (char*)d_ws;
    size_t off = 0;
    auto alloc = [&](size_t bytes) -> char* {
        char* p = ws + off;
        off = (off + bytes + 255) & ~(size_t)255;
        return p;
    };
    float*       wsum  = (float*)alloc(2 * sizeof(float));
    float*       wscl  = (float*)alloc(2 * sizeof(float));
    float*       a1    = (float*)alloc(NTOK * sizeof(float));
    float*       a2    = (float*)alloc(NTOK * sizeof(float));
    signed char* w1q   = (signed char*)alloc(nW1);
    signed char* w2q   = (signed char*)alloc(nW2);
    signed char* xq    = (signed char*)alloc(nX);
    signed char* hq    = (signed char*)alloc(nH);
    float*       h     = (float*)alloc(nH * sizeof(float));
    (void)ws_size;

    zero_f32<<<1, 64, 0, stream>>>(wsum, 2);
    abs_sum_kernel<<<1024, 256, 0, stream>>>(w1, nW1, &wsum[0]);
    abs_sum_kernel<<<1024, 256, 0, stream>>>(w2, nW2, &wsum[1]);
    weight_scale_kernel<<<1, 64, 0, stream>>>(wsum, wscl, (float)nW1, (float)nW2);
    weight_quant_kernel<<<8192, 256, 0, stream>>>(w1, w1q, &wscl[0], nW1 / 4);
    weight_quant_kernel<<<8192, 256, 0, stream>>>(w2, w2q, &wscl[1], nW2 / 4);

    rms_act_quant_kernel<<<NTOK, 256, 0, stream>>>(x, xq, a1, DIM);

    {
        dim3 grid(INNER / 128, NTOK / 128);
        gemm_i8_wmma_kernel<true><<<grid, 256, 0, stream>>>(
            xq, w1q, a1, &wscl[0], h, NTOK, INNER, DIM);
    }

    rms_act_quant_kernel<<<NTOK, 256, 0, stream>>>(h, hq, a2, INNER);

    {
        dim3 grid(DIM / 128, NTOK / 128);
        gemm_i8_wmma_kernel<false><<<grid, 256, 0, stream>>>(
            hq, w2q, a2, &wscl[1], out, NTOK, DIM, INNER);
    }
}
